// HeteroMAGNet_7121055776903
// MI455X (gfx1250) — matrix operand order repaired and checked by reading, not compiled, not run on hardware
//
#include <hip/hip_runtime.h>
#include <hip/hip_bf16.h>
#include <cstdint>

typedef __attribute__((ext_vector_type(16))) _Float16 v16h;
typedef __attribute__((ext_vector_type(8)))  _Float16 v8h;
typedef __attribute__((ext_vector_type(8)))  float    v8f;

// ---------------------------------------------------------------------------
// Weight transpose + fp32 -> f16 convert:  Bt[n][k] = W[k][n]  (zero-padded
// to Npad rows so WMMA column tiles never need EXEC masking before the MMA).
// ---------------------------------------------------------------------------
__global__ void convert_wT_kernel(const float* __restrict__ W, _Float16* __restrict__ Bt,
                                  int K, int N, int Npad) {
    int idx = blockIdx.x * blockDim.x + threadIdx.x;
    if (idx >= Npad * K) return;
    int n = idx / K;
    int k = idx - n * K;
    float v = (n < N) ? W[(size_t)k * N + n] : 0.0f;
    Bt[idx] = (_Float16)v;
}

// ---------------------------------------------------------------------------
// Register-blocked WMMA GEMM:  C(MxN) = act(A(MxK,lda) * B(KxN) + bias)
//   - one 16x(SUBT*16) output strip per wave; SUBT accumulators,
//     SUBT WMMAs per 32-K step, A fragment loaded/converted ONCE per K step
//   - K compile-time, fully unrolled: all A/B addresses are base+immediate ->
//     clause-able loads that stay in flight across the WMMA string
//   - AH: A is f16 (2x b128 per fragment, no cvts)  else f32 (4x b128 + cvts)
//   - CH: store C as f16 (halves traffic for GEMM-only consumers)
//   - Bt[n][k] f16, n padded to a multiple of SUBT*16 (zero rows)
//   - A/B use contiguous k = 16*half + i in both operands; f32 C/D per CDNA5 ISA
//   - SELF: additionally writes  C*dinv[row]^2 + bias2  into out2 (GCN fusion)
// ---------------------------------------------------------------------------
template<int K, int SUBT, bool RELU, bool BIAS, bool SELF, bool AH, bool CH>
__global__ void gemm_wmma_f16(const void* __restrict__ Ap, int lda, int M,
                              const _Float16* __restrict__ Bt,
                              const float* __restrict__ bias,
                              int N, void* __restrict__ Cp, int ldc,
                              float* __restrict__ out2, const float* __restrict__ dinv,
                              const float* __restrict__ bias2, int ldo) {
    constexpr int STRIPC = SUBT * 16;
    const int nstrips = (N + STRIPC - 1) / STRIPC;
    const int mtiles  = (M + 15) >> 4;
    const int wave    = blockIdx.x * (blockDim.x >> 5) + (threadIdx.x >> 5);
    if (wave >= mtiles * nstrips) return;          // uniform per-wave exit
    const int mt = wave / nstrips;
    const int ns = wave - mt * nstrips;

    const int lane = threadIdx.x & 31;
    const int half = lane >> 4;
    const int l15  = lane & 15;

    const int    mrow = mt * 16 + l15;
    const size_t arow = (size_t)((mrow < M) ? mrow : (M - 1)) * (size_t)lda;
    const _Float16* __restrict__ brow = Bt + (size_t)(ns * STRIPC + l15) * (size_t)K;

    v8f acc[SUBT] = {};
#pragma unroll
    for (int kt = 0; kt < K; kt += 32) {
        const int k0 = kt + half * 16;
        v16h a;
        if (AH) {
            const _Float16* Ahp = (const _Float16*)Ap;
            v8h lo = *(const v8h*)(Ahp + arow + k0);        // K = k0..k0+7   (16B)
            v8h hi = *(const v8h*)(Ahp + arow + k0 + 8);    // K = k0+8..+15  (16B)
#pragma unroll
            for (int i = 0; i < 8; ++i) { a[i] = lo[i]; a[8 + i] = hi[i]; }
        } else {
            const float* Afp = (const float*)Ap;
            float af[16];
            *(float4*)(af + 0)  = *(const float4*)(Afp + arow + k0 + 0);
            *(float4*)(af + 4)  = *(const float4*)(Afp + arow + k0 + 4);
            *(float4*)(af + 8)  = *(const float4*)(Afp + arow + k0 + 8);
            *(float4*)(af + 12) = *(const float4*)(Afp + arow + k0 + 12);
#pragma unroll
            for (int i = 0; i < 16; ++i) a[i] = (_Float16)af[i];
        }

        v16h b[SUBT];
#pragma unroll
        for (int s = 0; s < SUBT; ++s)
            b[s] = *(const v16h*)(brow + (size_t)s * (16 * K) + kt + half * 16);

#pragma unroll
        for (int s = 0; s < SUBT; ++s)
            acc[s] = __builtin_amdgcn_wmma_f32_16x16x32_f16(
                         false, a, false, b[s], (short)0, acc[s], false, false);
    }

    // self-term coefficients (per output row), only for the fused GCN path
    float selfc[8];
    if (SELF) {
#pragma unroll
        for (int j = 0; j < 8; ++j) {
            int row = mt * 16 + half * 8 + j;
            float di = dinv[(row < M) ? row : (M - 1)];
            selfc[j] = di * di;
        }
    }

#pragma unroll
    for (int s = 0; s < SUBT; ++s) {
        const int col = ns * STRIPC + s * 16 + l15;
        if (col < N) {
            const float bv  = BIAS ? bias[col]  : 0.0f;
            const float bv2 = SELF ? bias2[col] : 0.0f;
#pragma unroll
            for (int j = 0; j < 8; ++j) {
                const int row = mt * 16 + half * 8 + j;
                if (row < M) {
                    float v = acc[s][j] + bv;
                    if (RELU) v = fmaxf(v, 0.0f);
                    if (CH) ((_Float16*)Cp)[(size_t)row * ldc + col] = (_Float16)v;
                    else    ((float*)Cp)[(size_t)row * ldc + col] = v;
                    if (SELF) out2[(size_t)row * ldo + col] = v * selfc[j] + bv2;
                }
            }
        }
    }
}

// host-side typed launcher
template<int K, int SUBT, bool RELU, bool BIAS, bool SELF, bool AH, bool CH>
static void launch_gemm(hipStream_t stream, const void* A, int lda, int M,
                        const _Float16* Bt, const float* bias, int Nc, void* C, int ldc,
                        float* out2 = nullptr, const float* dinv = nullptr,
                        const float* bias2 = nullptr, int ldo = 0) {
    const int nstrips = (Nc + SUBT * 16 - 1) / (SUBT * 16);
    const int tiles   = ((M + 15) / 16) * nstrips;
    dim3 grid((tiles + 7) / 8), block(256);        // 8 waves per block
    gemm_wmma_f16<K, SUBT, RELU, BIAS, SELF, AH, CH><<<grid, block, 0, stream>>>(
        A, lda, M, Bt, bias, Nc, C, ldc, out2, dinv, bias2, ldo);
}

// ---------------------------------------------------------------------------
// Elementwise / graph kernels
// ---------------------------------------------------------------------------
__global__ void fill_kernel(float* __restrict__ p, float v, int n) {
    int i = blockIdx.x * blockDim.x + threadIdx.x;
    if (i < n) p[i] = v;
}

__global__ void deg_accum_kernel(const long long* __restrict__ dst,
                                 float* __restrict__ deg, int E) {
    int e = blockIdx.x * blockDim.x + threadIdx.x;
    if (e < E) atomicAdd(&deg[(int)dst[e]], 1.0f);
}

__global__ void rsqrt_kernel(const float* __restrict__ deg, float* __restrict__ dinv, int n) {
    int i = blockIdx.x * blockDim.x + threadIdx.x;
    if (i < n) dinv[i] = rsqrtf(deg[i]);
}

__global__ void select_h16_kernel(const _Float16* __restrict__ h0, const _Float16* __restrict__ h1,
                                  const long long* __restrict__ node_type,
                                  _Float16* __restrict__ h, int total) {
    int i = blockIdx.x * blockDim.x + threadIdx.x;
    if (i < total) {
        int node = i >> 7;                       // 128 features/node
        h[i] = (node_type[node] == 0) ? h0[i] : h1[i];
    }
}

// one wave per edge; lane covers 4 features -> coalesced f16 gather + f32 atomics
__global__ void gcn_scatter_kernel(const _Float16* __restrict__ zw,
                                   const long long* __restrict__ src,
                                   const long long* __restrict__ dst,
                                   const float* __restrict__ dinv,
                                   float* __restrict__ out, int ldo, int E) {
    int e    = (blockIdx.x * blockDim.x + threadIdx.x) >> 5;
    int lane = threadIdx.x & 31;
    if (e >= E) return;
    int s = (int)src[e], d = (int)dst[e];
    float c = dinv[s] * dinv[d];
    const _Float16* zr = zw + (size_t)s * 128;
    float* orow        = out + (size_t)d * ldo;
#pragma unroll
    for (int j = 0; j < 4; ++j) {
        int f = j * 32 + lane;
        atomicAdd(&orow[f], (float)zr[f] * c);
    }
}

__global__ void relu_cols_kernel(float* __restrict__ out, int ldo, int total) {
    int i = blockIdx.x * blockDim.x + threadIdx.x;
    if (i < total) {
        int node = i >> 7, f = i & 127;
        float* p = out + (size_t)node * ldo + f;
        *p = fmaxf(*p, 0.0f);
    }
}

// ---------------------------------------------------------------------------
// Host launcher
// ---------------------------------------------------------------------------
extern "C" void kernel_launch(void* const* d_in, const int* in_sizes, int n_in,
                              void* d_out, int out_size, void* d_ws, size_t ws_size,
                              hipStream_t stream) {
    const float*     x         = (const float*)d_in[0];
    const long long* edge      = (const long long*)d_in[1];
    const long long* node_type = (const long long*)d_in[2];
    const float* enc0_w1 = (const float*)d_in[3];
    const float* enc0_b1 = (const float*)d_in[4];
    const float* enc0_w2 = (const float*)d_in[5];
    const float* enc0_b2 = (const float*)d_in[6];
    const float* enc1_w1 = (const float*)d_in[7];
    const float* enc1_b1 = (const float*)d_in[8];
    const float* enc1_w2 = (const float*)d_in[9];
    const float* enc1_b2 = (const float*)d_in[10];
    const float* gcn_w[3] = {(const float*)d_in[11], (const float*)d_in[13], (const float*)d_in[15]};
    const float* gcn_b[3] = {(const float*)d_in[12], (const float*)d_in[14], (const float*)d_in[16]};
    const float* act_w1 = (const float*)d_in[17];
    const float* act_b1 = (const float*)d_in[18];
    const float* act_w2 = (const float*)d_in[19];
    const float* act_b2 = (const float*)d_in[20];

    const int N  = in_sizes[0] / 128;   // 50000 nodes
    const int E  = in_sizes[1] / 2;     // 800000 edges
    const int NA = out_size / 10;       // 25000 agents
    const long long* srcI = edge;
    const long long* dstI = edge + E;

    // ---- workspace carve (256B-aligned chunks) ----
    char*  ws  = (char*)d_ws;
    size_t off = 0;
    auto take = [&](size_t bytes) -> void* {
        off = (off + 255) & ~(size_t)255;
        void* p = ws + off;
        off += bytes;
        return p;
    };
    _Float16* wt_enc0_1 = (_Float16*)take((size_t)128 * 128 * 2);
    _Float16* wt_enc0_2 = (_Float16*)take((size_t)128 * 128 * 2);
    _Float16* wt_enc1_1 = (_Float16*)take((size_t)128 * 64  * 2);
    _Float16* wt_enc1_2 = (_Float16*)take((size_t)128 * 128 * 2);
    _Float16* wt_gcn[3];
    for (int l = 0; l < 3; ++l) wt_gcn[l] = (_Float16*)take((size_t)128 * 128 * 2);
    _Float16* wt_act1 = (_Float16*)take((size_t)256 * 384 * 2);
    _Float16* wt_act2 = (_Float16*)take((size_t)64  * 256 * 2);   // N 10 -> 64 pad
    float* deg  = (float*)take((size_t)N * 4);
    float* dinv = (float*)take((size_t)N * 4);
    // f16 activation buffers
    _Float16* B1 = (_Float16*)take((size_t)N * 128 * 2);   // t0 / t1 / h
    _Float16* B2 = (_Float16*)take((size_t)N * 128 * 2);   // h0 / zw
    _Float16* B3 = (_Float16*)take((size_t)N * 128 * 2);   // h1 / actor hidden (25000x256)
    float* xcat  = (float*)take((size_t)N * 384 * 4);      // [o1 | o2 | o3] (f32: atomic dest)
    (void)ws_size; (void)n_in;

    auto cdiv = [](int a, int b) { return (a + b - 1) / b; };

    auto convert = [&](const float* W, _Float16* Bt, int K, int Ncols, int Npad) {
        int total = Npad * K;
        convert_wT_kernel<<<cdiv(total, 256), 256, 0, stream>>>(W, Bt, K, Ncols, Npad);
    };

    // ---- weight prep (tiny: ~0.4 MB total) ----
    convert(enc0_w1, wt_enc0_1, 128, 128, 128);
    convert(enc0_w2, wt_enc0_2, 128, 128, 128);
    convert(enc1_w1, wt_enc1_1, 64,  128, 128);
    convert(enc1_w2, wt_enc1_2, 128, 128, 128);
    for (int l = 0; l < 3; ++l) convert(gcn_w[l], wt_gcn[l], 128, 128, 128);
    convert(act_w1, wt_act1, 384, 256, 256);
    convert(act_w2, wt_act2, 256, 10, 64);                 // pad N 10 -> 64 with zeros

    // ---- degree normalization ----
    fill_kernel<<<cdiv(N, 256), 256, 0, stream>>>(deg, 1.0f, N);
    deg_accum_kernel<<<cdiv(E, 256), 256, 0, stream>>>(dstI, deg, E);
    rsqrt_kernel<<<cdiv(N, 256), 256, 0, stream>>>(deg, dinv, N);

    // ---- heterogeneous encoders (intermediates stored f16) ----
    //                K    S  RELU  BIAS  SELF  AH    CH
    launch_gemm<128, 8, true,  true,  false, false, true >(stream, x,  128, N, wt_enc0_1, enc0_b1, 128, B1, 128); // t0
    launch_gemm<128, 8, false, true,  false, true,  true >(stream, B1, 128, N, wt_enc0_2, enc0_b2, 128, B2, 128); // h0
    launch_gemm<64,  8, true,  true,  false, false, true >(stream, x,  128, N, wt_enc1_1, enc1_b1, 128, B1, 128); // t1
    launch_gemm<128, 8, false, true,  false, true,  true >(stream, B1, 128, N, wt_enc1_2, enc1_b2, 128, B3, 128); // h1
    select_h16_kernel<<<cdiv(N * 128, 256), 256, 0, stream>>>(B2, B3, node_type, B1, N * 128);                    // h

    // ---- 3 GCN layers, outputs written as column blocks of xcat (f32) ----
    for (int l = 0; l < 3; ++l) {
        float* outcol = xcat + (size_t)l * 128;
        // zw = A*W (f16)  and  xcat_col = zw*dinv^2 + b   (fused epilogue)
        if (l == 0)
            launch_gemm<128, 8, false, false, true, true,  true>(stream, B1, 128, N, wt_gcn[l], nullptr, 128,
                                                                 B2, 128, outcol, dinv, gcn_b[l], 384);
        else
            launch_gemm<128, 8, false, false, true, false, true>(stream, xcat + (size_t)(l - 1) * 128, 384, N,
                                                                 wt_gcn[l], nullptr, 128,
                                                                 B2, 128, outcol, dinv, gcn_b[l], 384);
        gcn_scatter_kernel<<<cdiv(E, 8), 256, 0, stream>>>(B2, srcI, dstI, dinv, outcol, 384, E);
        relu_cols_kernel<<<cdiv(N * 128, 256), 256, 0, stream>>>(outcol, 384, N * 128);
    }

    // ---- actor head on first NA rows (concat is free: xcat row = [o1|o2|o3]) ----
    launch_gemm<384, 8, true,  true, false, false, true >(stream, xcat, 384, NA, wt_act1, act_b1, 256, B3, 256); // hidden
    launch_gemm<256, 4, false, true, false, true,  false>(stream, B3, 256, NA, wt_act2, act_b2, 10,
                                                          d_out, 10);                                            // q
}
